// ABCAttention_68204080660975
// MI455X (gfx1250) — compile-verified, matrix operand order checked by reading, not run on hardware
//
#include <hip/hip_runtime.h>
#include <hip/hip_bf16.h>
#include <math.h>
#include <stdint.h>

// ---------------- CDNA5 WMMA plumbing (gfx1250, wave32) ----------------
typedef __attribute__((ext_vector_type(16))) _Float16 v16h;
typedef __attribute__((ext_vector_type(8)))  _Float16 h8;
typedef __attribute__((ext_vector_type(4)))  _Float16 h4;
typedef __attribute__((ext_vector_type(4)))  float    f4;
typedef __attribute__((ext_vector_type(8)))  float    v8f;

__device__ __forceinline__ v8f wmma32(v16h a, v16h b, v8f c) {
  return __builtin_amdgcn_wmma_f32_16x16x32_f16(
      false, a, false, b, (short)0, c, false, false);
}

// 16x32 f16 operand fragment from LDS (ISA 7.12.2 16-bit A layout; B operands
// staged transposed [Ncol][K] so the same loader applies).
__device__ __forceinline__ v16h ldfrag(const _Float16* base, int lane, int stride) {
  const int r  = lane & 15;
  const int kb = (lane >> 4) << 3;
  const _Float16* p = base + r * stride;
  h8 lo = *(const h8*)(p + kb);
  h8 hi = *(const h8*)(p + 16 + kb);
  v16h out;
#pragma unroll
  for (int i = 0; i < 8; ++i) { out[i] = lo[i]; out[i + 8] = hi[i]; }
  return out;
}

// CDNA5 async global->LDS copy of one 16-byte chunk per lane (ASYNCcnt).
__device__ __forceinline__ void async_cp16(uint32_t lds_off, const void* gptr) {
  asm volatile("global_load_async_to_lds_b128 %0, %1, off"
               : : "v"(lds_off), "v"(gptr) : "memory");
}
#define ASYNC_WAIT0() asm volatile("s_wait_asynccnt 0x0" ::: "memory")
#define ASYNC_WAIT6() asm volatile("s_wait_asynccnt 0x6" ::: "memory")

__device__ __forceinline__ uint32_t lds_addr(const void* p) {
  return (uint32_t)(uintptr_t)p;   // low 32 bits of generic ptr = LDS offset
}

// Problem constants
#define BB   2
#define TT   2048
#define DD   1024
#define HH   4
#define DK   256
#define MM   64
#define ROWS (BB * TT)          // 4096
#define BH   (BB * HH)          // 8

// ---------------- one-shot f32 -> f16 conversions ---------------------------
__global__ void cvt_f16_kernel(const float* __restrict__ src, _Float16* __restrict__ dst) {
  int i = blockIdx.x * blockDim.x + threadIdx.x;  // n/4 threads
  f4 v = ((const f4*)src)[i];
  h4 h;
#pragma unroll
  for (int j = 0; j < 4; ++j) h[j] = (_Float16)v[j];
  ((h4*)dst)[i] = h;
}

// Wt[n][k] = (f16) W[k][n]   (1024 x 1024)
__global__ void cvt_wt_kernel(const float* __restrict__ W, _Float16* __restrict__ Wt) {
  int gid = blockIdx.x * blockDim.x + threadIdx.x;
  int k = gid >> 10, n = gid & 1023;
  Wt[(size_t)n * DD + k] = (_Float16)W[(size_t)k * DD + n];
}

// ---------------- WMMA GEMM, f16 in, async double-buffered ------------------
// Block tile 128x64, 4 waves, each wave 32x64 (8 WMMAs / k-step).
// mode 0: f32 row-major; mode 1: f16 scatter (B,H,T,DK); mode 2: f16 (B,H,DK,T)
__device__ __forceinline__ void stage_tiles(const _Float16* Ah, const _Float16* Wt,
                                            int K, int m0, int n0, int kk,
                                            uint32_t asb, uint32_t bsb, int tid) {
#pragma unroll
  for (int c = 0; c < 4; ++c) {            // A: 128 rows x 4 x 16B
    int e = tid + 128 * c;
    int row = e >> 2, sub = e & 3;
    async_cp16(asb + (uint32_t)(row * 32 + sub * 8) * 2,
               Ah + (size_t)(m0 + row) * K + kk + sub * 8);
  }
#pragma unroll
  for (int c = 0; c < 2; ++c) {            // B: 64 rows x 4 x 16B
    int e = tid + 128 * c;
    int row = e >> 2, sub = e & 3;
    async_cp16(bsb + (uint32_t)(row * 32 + sub * 8) * 2,
               Wt + (size_t)(n0 + row) * K + kk + sub * 8);
  }
}

__global__ void gemm_wmma_kernel(const _Float16* __restrict__ Ah,
                                 const _Float16* __restrict__ Wt,
                                 float* __restrict__ Cf,
                                 _Float16* __restrict__ Ch,
                                 int K, int Ncols, int mode) {
  __shared__ __align__(16) _Float16 As[2][128 * 32];   // 16 KB
  __shared__ __align__(16) _Float16 Bs[2][64 * 32];    //  8 KB
  const int tid  = threadIdx.x;            // 0..127 (4 waves)
  const int wave = tid >> 5, lane = tid & 31;
  const int m0 = blockIdx.y * 128, n0 = blockIdx.x * 64;
  const uint32_t as0 = lds_addr(&As[0][0]), as1 = lds_addr(&As[1][0]);
  const uint32_t bs0 = lds_addr(&Bs[0][0]), bs1 = lds_addr(&Bs[1][0]);
  v8f acc[2][4] = {};

  auto compute = [&](int cur) {
    const _Float16* Ab = &As[cur][0];
    const _Float16* Bb = &Bs[cur][0];
    v16h a0 = ldfrag(Ab + (wave * 32) * 32, lane, 32);
    v16h a1 = ldfrag(Ab + (wave * 32 + 16) * 32, lane, 32);
#pragma unroll
    for (int n = 0; n < 4; ++n) {
      v16h b = ldfrag(Bb + (n * 16) * 32, lane, 32);
      acc[0][n] = wmma32(a0, b, acc[0][n]);
      acc[1][n] = wmma32(a1, b, acc[1][n]);
    }
  };

  stage_tiles(Ah, Wt, K, m0, n0, 0, as0, bs0, tid);   // prologue -> buf 0
  int kk = 0;
  for (; kk + 32 < K; kk += 32) {                     // branchless steady state
    const int cur = (kk >> 5) & 1;
    stage_tiles(Ah, Wt, K, m0, n0, kk + 32, cur ? as0 : as1, cur ? bs0 : bs1, tid);
    ASYNC_WAIT6();        // in-order completion: previous tile's 6 copies done
    __syncthreads();
    compute(cur);
    __syncthreads();
  }
  ASYNC_WAIT0();                                      // peeled last tile
  __syncthreads();
  compute((kk >> 5) & 1);

  const int hi = lane >> 4, lo = lane & 15;
#pragma unroll
  for (int i = 0; i < 2; ++i)
#pragma unroll
    for (int n = 0; n < 4; ++n)
#pragma unroll
      for (int r = 0; r < 8; ++r) {
        int grow = m0 + wave * 32 + i * 16 + r + 8 * hi;
        int gcol = n0 + n * 16 + lo;
        float val = acc[i][n][r];
        if (mode == 0) {
          Cf[(size_t)grow * Ncols + gcol] = val;
        } else {
          int b = grow >> 11, t = grow & (TT - 1);
          int h = gcol >> 8,  d = gcol & (DK - 1);
          if (mode == 1)
            Ch[(((size_t)(b * HH + h)) * TT + t) * DK + d] = (_Float16)val;
          else
            Ch[(((size_t)(b * HH + h)) * DK + d) * TT + t] = (_Float16)val;
        }
      }
}

// ---------------- Score path ------------------------------------------------
__global__ void score1_kernel(const float* __restrict__ x,
                              const float* __restrict__ w1,
                              float* __restrict__ t1) {
  int gid = blockIdx.x * blockDim.x + threadIdx.x;       // ROWS*16
  int row = gid >> 4, j = gid & 15;
  float acc = 0.f;
  for (int i = 0; i < DD; ++i) acc += x[(size_t)row * DD + i] * w1[i * 16 + j];
  t1[row * 16 + j] = acc;
}

__global__ void score2_kernel(const float* __restrict__ t1,
                              const float* __restrict__ w2,
                              const float* __restrict__ b2,
                              float* __restrict__ w32,
                              _Float16* __restrict__ wh,
                              int transposed) {
  int gid = blockIdx.x * blockDim.x + threadIdx.x;       // ROWS*256
  int row = gid >> 8, col = gid & 255;
  float acc = b2[col];
#pragma unroll
  for (int i = 0; i < 16; ++i) acc += t1[row * 16 + i] * w2[i * (HH * MM) + col];
  acc = fminf(32.f, fmaxf(-32.f, acc));
  float e = __expf(acc);
  int b = row >> 11, t = row & (TT - 1);
  int h = col >> 6,  m = col & (MM - 1);
  int bh = b * HH + h;
  w32[((size_t)bh * TT + t) * MM + m] = e;
  _Float16 ec = (_Float16)fminf(e, 60000.f);
  if (transposed) wh[((size_t)bh * MM + m) * TT + t] = ec;
  else            wh[((size_t)bh * TT + t) * MM + m] = ec;
}

__global__ void cumsum_kernel(const float* __restrict__ w, float* __restrict__ z) {
  int gid = blockIdx.x * blockDim.x + threadIdx.x;       // BH*MM = 512
  if (gid >= BH * MM) return;
  int bh = gid >> 6, m = gid & (MM - 1);
  float run = 0.f;
  for (int t = 0; t < TT; ++t) {
    size_t idx = ((size_t)bh * TT + t) * MM + m;
    run += w[idx];
    z[idx] = run;
  }
}

// ---------------- RoPE in place on f16 (B,H,T,DK) ---------------------------
__global__ void rope_kernel(_Float16* __restrict__ p) {
  int gid = blockIdx.x * blockDim.x + threadIdx.x;       // BH*T*128
  int d  = gid & 127;
  int t  = (gid >> 7) & (TT - 1);
  int bh = gid >> 18;
  size_t base = ((size_t)bh * TT + t) * DK;
  float ang = (float)t * __expf(-logf(10000.f) * (float)d * (1.f / 128.f));
  float c = __cosf(ang), s = __sinf(ang);
  float x1 = (float)p[base + d], x2 = (float)p[base + 128 + d];
  p[base + d]       = (_Float16)(x1 * c - x2 * s);
  p[base + 128 + d] = (_Float16)(x2 * c + x1 * s);
}

// ---------------- Attention phase 1 -----------------------------------------
// 4 waves / 64 query rows per block; Ks/WKs staged once, shared by all waves.
__global__ void attn1_kernel(const _Float16* __restrict__ q,
                             const _Float16* __restrict__ k,
                             const _Float16* __restrict__ wkh,   // [bh][m][T]
                             const float* __restrict__ zk,
                             float* __restrict__ logits) {
  __shared__ __align__(16) _Float16 Qs[64 * DK];      // 32 KB
  __shared__ __align__(16) _Float16 Ks[32 * DK];      // 16 KB
  __shared__ __align__(16) _Float16 WKs[MM * 32];     //  4 KB
  __shared__ __align__(16) _Float16 Ss[4][16 * 32];   //  4 KB (per wave)
  const int tid  = threadIdx.x;
  const int wave = tid >> 5, lane = tid & 31;
  const int bh  = blockIdx.y;
  const int t0b = blockIdx.x * 64;
  const int wt0 = t0b + wave * 16;
  const size_t rowbase = (size_t)bh * TT;
  const int hi = lane >> 4, lo = lane & 15;
  const uint32_t qsb = lds_addr(Qs), ksb = lds_addr(Ks), wksb = lds_addr(WKs);

#pragma unroll
  for (int i = 0; i < 16; ++i) {           // Qs: 64 x 256 halves
    int e = tid + 128 * i; int r = e >> 5, sub = e & 31;
    async_cp16(qsb + (uint32_t)(r * DK + sub * 8) * 2,
               q + (rowbase + t0b + r) * DK + sub * 8);
  }
  v8f L[4] = {};

  for (int s0 = 0; s0 < t0b + 64; s0 += 32) {
#pragma unroll
    for (int i = 0; i < 8; ++i) {          // Ks: 32 x 256 halves
      int e = tid + 128 * i; int s = e >> 5, sub = e & 31;
      async_cp16(ksb + (uint32_t)(s * DK + sub * 8) * 2,
                 k + (rowbase + s0 + s) * DK + sub * 8);
    }
#pragma unroll
    for (int i = 0; i < 2; ++i) {          // WKs: 64(m) x 32(s) halves
      int e = tid + 128 * i; int m = e >> 2, sub = e & 3;
      async_cp16(wksb + (uint32_t)(m * 32 + sub * 8) * 2,
                 wkh + ((size_t)bh * MM + m) * TT + s0 + sub * 8);
    }
    ASYNC_WAIT0();
    __syncthreads();
    if (s0 < wt0 + 16) {                   // wave-uniform causal guard
      const _Float16* Qw = Qs + (wave * 16) * DK;
#pragma unroll
      for (int j = 0; j < 2; ++j) {
        v8f sacc = {};
#pragma unroll
        for (int dk = 0; dk < DK; dk += 32) {
          v16h a = ldfrag(Qw + dk, lane, DK);
          v16h b = ldfrag(Ks + (j * 16) * DK + dk, lane, DK);
          sacc = wmma32(a, b, sacc);
        }
        int scol = s0 + j * 16 + lo;
#pragma unroll
        for (int r = 0; r < 8; ++r) {
          int trow = wt0 + r + 8 * hi;
          float vv = (scol <= trow) ? sacc[r] : 0.f;
          Ss[wave][(r + 8 * hi) * 32 + j * 16 + lo] = (_Float16)vv;
        }
      }
      // per-wave LDS round-trip: DS ops are in-order within a wave
      v16h a = ldfrag(Ss[wave], lane, 32);
#pragma unroll
      for (int n = 0; n < 4; ++n) {
        v16h b = ldfrag(WKs + (n * 16) * 32, lane, 32);
        L[n] = wmma32(a, b, L[n]);
      }
    }
    __syncthreads();
  }

#pragma unroll
  for (int n = 0; n < 4; ++n)
#pragma unroll
    for (int r = 0; r < 8; ++r) {
      int t = wt0 + r + 8 * hi, m = n * 16 + lo;
      size_t idx = (rowbase + t) * MM + m;
      logits[idx] = L[n][r] * 0.0625f / zk[idx];
    }
}

// softmax over M=64 fused with /zv -> uh (f16, [bh][t][m])
__global__ void softmax_kernel(const float* __restrict__ lg,
                               const float* __restrict__ zv,
                               _Float16* __restrict__ uh) {
  int gid = blockIdx.x * blockDim.x + threadIdx.x;  // BH*T rows
  size_t base = (size_t)gid * MM;
  float mx = -1e30f;
  for (int m = 0; m < MM; ++m) mx = fmaxf(mx, lg[base + m]);
  float sum = 0.f;
  for (int m = 0; m < MM; ++m) sum += __expf(lg[base + m] - mx);
  float inv = 1.f / sum;
  for (int m = 0; m < MM; ++m) {
    float u = __expf(lg[base + m] - mx) * inv / zv[base + m];
    uh[base + m] = (_Float16)fminf(u, 60000.f);
  }
}

// ---------------- Attention phase 2: o = tril(u wv^T) @ v -------------------
// 4 waves / 64 query rows per block; WVs/Vt staged once, shared.
__global__ void attn2_kernel(const _Float16* __restrict__ uh,   // [bh][t][m]
                             const _Float16* __restrict__ wvh,  // [bh][t][m]
                             const _Float16* __restrict__ vt,   // [bh][d][T]
                             float* __restrict__ o) {
  __shared__ __align__(16) _Float16 Us[64 * MM];      //  8 KB
  __shared__ __align__(16) _Float16 WVs[32 * MM];     //  4 KB
  __shared__ __align__(16) _Float16 As16[4][16 * 32]; //  4 KB (per wave)
  __shared__ __align__(16) _Float16 Vt[DK * 32];      // 16 KB
  const int tid  = threadIdx.x;
  const int wave = tid >> 5, lane = tid & 31;
  const int bh  = blockIdx.y;
  const int t0b = blockIdx.x * 64;
  const int wt0 = t0b + wave * 16;
  const size_t rowbase = (size_t)bh * TT;
  const int hi = lane >> 4, lo = lane & 15;
  const uint32_t usb = lds_addr(Us), wvsb = lds_addr(WVs), vtb = lds_addr(Vt);

#pragma unroll
  for (int i = 0; i < 4; ++i) {            // Us: 64 x 64 halves
    int e = tid + 128 * i; int r = e >> 3, sub = e & 7;
    async_cp16(usb + (uint32_t)(r * MM + sub * 8) * 2,
               uh + (rowbase + t0b + r) * MM + sub * 8);
  }
  v8f O[16] = {};

  for (int s0 = 0; s0 < t0b + 64; s0 += 32) {
#pragma unroll
    for (int i = 0; i < 2; ++i) {          // WVs: 32 x 64 halves
      int e = tid + 128 * i; int s = e >> 3, sub = e & 7;
      async_cp16(wvsb + (uint32_t)(s * MM + sub * 8) * 2,
                 wvh + (rowbase + s0 + s) * MM + sub * 8);
    }
#pragma unroll
    for (int i = 0; i < 8; ++i) {          // Vt: 256(d) x 32(s) halves
      int e = tid + 128 * i; int d = e >> 2, sub = e & 3;
      async_cp16(vtb + (uint32_t)(d * 32 + sub * 8) * 2,
                 vt + ((size_t)bh * DK + d) * TT + s0 + sub * 8);
    }
    ASYNC_WAIT0();
    __syncthreads();
    if (s0 < wt0 + 16) {                   // wave-uniform causal guard
      const _Float16* Uw = Us + (wave * 16) * MM;
#pragma unroll
      for (int j = 0; j < 2; ++j) {
        v8f aacc = {};
#pragma unroll
        for (int mk = 0; mk < MM; mk += 32) {
          v16h a = ldfrag(Uw + mk, lane, MM);
          v16h b = ldfrag(WVs + (j * 16) * MM + mk, lane, MM);
          aacc = wmma32(a, b, aacc);
        }
        int scol = s0 + j * 16 + lo;
#pragma unroll
        for (int r = 0; r < 8; ++r) {
          int trow = wt0 + r + 8 * hi;
          float vv = (scol <= trow) ? aacc[r] : 0.f;
          As16[wave][(r + 8 * hi) * 32 + j * 16 + lo] = (_Float16)vv;
        }
      }
      v16h a2 = ldfrag(As16[wave], lane, 32);
#pragma unroll
      for (int n = 0; n < 16; ++n) {
        v16h b = ldfrag(Vt + (n * 16) * 32, lane, 32);
        O[n] = wmma32(a2, b, O[n]);
      }
    }
    __syncthreads();
  }

#pragma unroll
  for (int n = 0; n < 16; ++n)
#pragma unroll
    for (int r = 0; r < 8; ++r) {
      int t = wt0 + r + 8 * hi, d = n * 16 + lo;
      o[(rowbase + t) * DK + d] = O[n][r];
    }
}

// ---------------- RMS-norm + SiLU gate -> f16 [row][col] --------------------
__global__ void gate_kernel(const float* __restrict__ o,
                            const float* __restrict__ g,
                            const float* __restrict__ gw,
                            _Float16* __restrict__ ogh) {
  int gid = blockIdx.x * blockDim.x + threadIdx.x;    // BH*T rows
  int bh = gid >> 11, t = gid & (TT - 1);
  int b = bh >> 2, h = bh & 3;
  size_t obase = ((size_t)bh * TT + t) * DK;
  size_t gbase = ((size_t)(b * TT + t)) * DD + h * DK;
  float ss = 0.f;
  for (int d = 0; d < DK; ++d) { float vv = o[obase + d]; ss += vv * vv; }
  float rms = rsqrtf(ss * (1.f / (float)DK) + 1e-5f);
  for (int d = 0; d < DK; ++d) {
    float on  = o[obase + d] * rms * gw[d];
    float gv  = g[gbase + d];
    float sig = 1.f / (1.f + __expf(-gv));
    ogh[gbase + d] = (_Float16)(on * gv * sig);
  }
}

// ---------------- Launch ----------------------------------------------------
extern "C" void kernel_launch(void* const* d_in, const int* in_sizes, int n_in,
                              void* d_out, int out_size, void* d_ws, size_t ws_size,
                              hipStream_t stream) {
  const float* x     = (const float*)d_in[0];
  const float* Wq    = (const float*)d_in[1];
  const float* Wk    = (const float*)d_in[2];
  const float* Wv    = (const float*)d_in[3];
  const float* Wg    = (const float*)d_in[4];
  const float* Wo    = (const float*)d_in[5];
  const float* sk_w1 = (const float*)d_in[6];
  const float* sk_w2 = (const float*)d_in[7];
  const float* sk_b2 = (const float*)d_in[8];
  const float* sv_w1 = (const float*)d_in[9];
  const float* sv_w2 = (const float*)d_in[10];
  const float* sv_b2 = (const float*)d_in[11];
  const float* gnw   = (const float*)d_in[12];
  float* out = (float*)d_out;

  char* w = (char*)d_ws;
  auto take = [&](size_t bytes) { char* p = w; w += (bytes + 255) & ~(size_t)255; return p; };
  const size_t N_BTD  = (size_t)ROWS * DD;
  const size_t N_BHTD = (size_t)BH * TT * DK;
  const size_t N_BHTM = (size_t)BH * TT * MM;

  _Float16* xh  = (_Float16*)take(N_BTD * 2);
  _Float16* Wtq = (_Float16*)take((size_t)DD * DD * 2);
  _Float16* Wtk = (_Float16*)take((size_t)DD * DD * 2);
  _Float16* Wtv = (_Float16*)take((size_t)DD * DD * 2);
  _Float16* Wtg = (_Float16*)take((size_t)DD * DD * 2);
  _Float16* Wto = (_Float16*)take((size_t)DD * DD * 2);
  _Float16* q_h = (_Float16*)take(N_BHTD * 2);
  _Float16* k_h = (_Float16*)take(N_BHTD * 2);
  _Float16* v_t = (_Float16*)take(N_BHTD * 2);   // (B,H,DK,T)
  float*    gbuf= (float*)   take(N_BTD * 4);
  float*    t1k = (float*)   take((size_t)ROWS * 16 * 4);
  float*    t1v = (float*)   take((size_t)ROWS * 16 * 4);
  float*    wk32= (float*)   take(N_BHTM * 4);
  float*    wv32= (float*)   take(N_BHTM * 4);
  float*    zk  = (float*)   take(N_BHTM * 4);
  float*    zv  = (float*)   take(N_BHTM * 4);
  _Float16* wkh = (_Float16*)take(N_BHTM * 2);   // [bh][m][T]
  _Float16* wvh = (_Float16*)take(N_BHTM * 2);   // [bh][t][m]
  float*    lg  = (float*)   take(N_BHTM * 4);
  _Float16* uh  = (_Float16*)take(N_BHTM * 2);
  float*    o_a = (float*)   take(N_BHTD * 4);
  _Float16* ogh = (_Float16*)take(N_BTD * 2);

  cvt_f16_kernel<<<(N_BTD / 4) / 256, 256, 0, stream>>>(x, xh);
  cvt_wt_kernel<<<(DD * DD) / 256, 256, 0, stream>>>(Wq, Wtq);
  cvt_wt_kernel<<<(DD * DD) / 256, 256, 0, stream>>>(Wk, Wtk);
  cvt_wt_kernel<<<(DD * DD) / 256, 256, 0, stream>>>(Wv, Wtv);
  cvt_wt_kernel<<<(DD * DD) / 256, 256, 0, stream>>>(Wg, Wtg);
  cvt_wt_kernel<<<(DD * DD) / 256, 256, 0, stream>>>(Wo, Wto);

  dim3 gemm_grid(DD / 64, ROWS / 128);           // (16, 32)
  gemm_wmma_kernel<<<gemm_grid, 128, 0, stream>>>(xh, Wtq, nullptr, q_h, DD, DD, 1);
  gemm_wmma_kernel<<<gemm_grid, 128, 0, stream>>>(xh, Wtk, nullptr, k_h, DD, DD, 1);
  gemm_wmma_kernel<<<gemm_grid, 128, 0, stream>>>(xh, Wtv, nullptr, v_t, DD, DD, 2);
  gemm_wmma_kernel<<<gemm_grid, 128, 0, stream>>>(xh, Wtg, gbuf, nullptr, DD, DD, 0);

  score1_kernel<<<(ROWS * 16) / 256, 256, 0, stream>>>(x, sk_w1, t1k);
  score1_kernel<<<(ROWS * 16) / 256, 256, 0, stream>>>(x, sv_w1, t1v);
  score2_kernel<<<(ROWS * 256) / 256, 256, 0, stream>>>(t1k, sk_w2, sk_b2, wk32, wkh, 1);
  score2_kernel<<<(ROWS * 256) / 256, 256, 0, stream>>>(t1v, sv_w2, sv_b2, wv32, wvh, 0);
  cumsum_kernel<<<2, 256, 0, stream>>>(wk32, zk);
  cumsum_kernel<<<2, 256, 0, stream>>>(wv32, zv);

  rope_kernel<<<(BH * TT * 128) / 256, 256, 0, stream>>>(q_h);
  rope_kernel<<<(BH * TT * 128) / 256, 256, 0, stream>>>(k_h);

  dim3 attn_grid(TT / 64, BH);                   // (32, 8), 4 waves/block
  attn1_kernel<<<attn_grid, 128, 0, stream>>>(q_h, k_h, wkh, zk, lg);
  softmax_kernel<<<(BH * TT) / 256, 256, 0, stream>>>(lg, zv, uh);
  attn2_kernel<<<attn_grid, 128, 0, stream>>>(uh, wvh, v_t, o_a);

  gate_kernel<<<(BH * TT) / 256, 256, 0, stream>>>(o_a, gbuf, gnw, ogh);
  gemm_wmma_kernel<<<gemm_grid, 128, 0, stream>>>(ogh, Wto, out, nullptr, DD, DD, 0);

  (void)in_sizes; (void)n_in; (void)out_size; (void)ws_size;
}
